// ComplexSelfAttention_43662637532026
// MI455X (gfx1250) — matrix operand order repaired and checked by reading, not compiled, not run on hardware
//
#include <hip/hip_runtime.h>
#include <hip/hip_bf16.h>

#define BB   8
#define CC   256
#define NN   2304        // 48*48
#define KK2  512         // 2*C
#define DQKV 1536        // 6*C
#define BN_EPS 1e-5f

typedef __bf16 bf16;
typedef __attribute__((ext_vector_type(16))) __bf16 v16bf;
typedef __attribute__((ext_vector_type(8)))  __bf16 v8bf;
typedef __attribute__((ext_vector_type(8)))  float  v8f;

// ---------------- CDNA5 async global->LDS copy (16 bytes / lane) --------
static __device__ inline void async_copy16(void* lds_dst, const void* gsrc) {
  unsigned l = (unsigned)(uintptr_t)lds_dst;              // low 32b = LDS offset
  unsigned long long g = (unsigned long long)(uintptr_t)gsrc;
  asm volatile("global_load_async_to_lds_b128 %0, %1, off"
               :: "v"(l), "v"(g) : "memory");
}
static __device__ inline void async_wait0() {
  asm volatile("s_wait_asynccnt 0x0" ::: "memory");
}

// ---------------- WMMA fragment loaders (bf16, 16x16x32) ----------------
// A-matrix 16x32: lane L -> row m=L&15 ; k-chunks at (L>>4)*8 and +16.
static __device__ inline v16bf frag_a(const bf16* __restrict__ base, int lane) {
  int m  = lane & 15;
  int k0 = (lane >> 4) * 8;
  const bf16* p0 = base + m * 32 + k0;
  v8bf lo = *(const v8bf*)(p0);
  v8bf hi = *(const v8bf*)(p0 + 16);
  v16bf f;
#pragma unroll
  for (int i = 0; i < 8; ++i) { f[i] = lo[i]; f[i + 8] = hi[i]; }
  return f;
}
// B-matrix 32x16 loaded from row-major [n][k] tile: lane L -> col n=L&15,
// k-chunks at (L>>4)*8 and +16.
static __device__ inline v16bf frag_b(const bf16* __restrict__ base, int lane) {
  int n  = lane & 15;
  int k0 = (lane >> 4) * 8;
  const bf16* p0 = base + n * 32 + k0;
  v8bf lo = *(const v8bf*)(p0);
  v8bf hi = *(const v8bf*)(p0 + 16);
  v16bf f;
#pragma unroll
  for (int i = 0; i < 8; ++i) { f[i] = lo[i]; f[i + 8] = hi[i]; }
  return f;
}

static __device__ inline v8f wmma_bf16(v16bf a, v16bf b, v8f c) {
  return __builtin_amdgcn_wmma_f32_16x16x32_bf16(false, a, false, b, (short)0, c, false, false);
}

// ---------------- K0a: pack Wcat [1536][512] bf16 + biasCat[1536] -------
__global__ void pack_w_kernel(const float* __restrict__ qwr, const float* __restrict__ qwi,
                              const float* __restrict__ qbr, const float* __restrict__ qbi,
                              const float* __restrict__ kwr, const float* __restrict__ kwi,
                              const float* __restrict__ kbr, const float* __restrict__ kbi,
                              const float* __restrict__ vwr, const float* __restrict__ vwi,
                              const float* __restrict__ vbr, const float* __restrict__ vbi,
                              bf16* __restrict__ Wcat, float* __restrict__ biasCat) {
  int idx = blockIdx.x * blockDim.x + threadIdx.x;
  if (idx >= DQKV * KK2) return;
  int d = idx / KK2, k = idx % KK2;
  int grp = d / KK2;         // 0=q,1=k,2=v   (group size == 512 == KK2)
  int r = d % KK2;
  int part = r >> 8, dd = r & 255;      // 0: real output row, 1: imag
  int half = k >> 8, cc = k & 255;      // 0: xr column,      1: xi
  const float *wr, *wi, *br, *bi;
  if (grp == 0)      { wr = qwr; wi = qwi; br = qbr; bi = qbi; }
  else if (grp == 1) { wr = kwr; wi = kwi; br = kbr; bi = kbi; }
  else               { wr = vwr; wi = vwi; br = vbr; bi = vbi; }
  float val;
  if (part == 0) val = (half == 0) ?  wr[dd * CC + cc] : -wi[dd * CC + cc];
  else           val = (half == 0) ?  wi[dd * CC + cc] :  wr[dd * CC + cc];
  Wcat[idx] = (bf16)val;
  if (k == 0) biasCat[d] = (part == 0) ? br[dd] : bi[dd];
}

// ---------------- K0b: pack X2 [B][N][512] bf16 (transpose of x) --------
__global__ void pack_x_kernel(const float* __restrict__ xr, const float* __restrict__ xi,
                              bf16* __restrict__ X2) {
  long idx = (long)blockIdx.x * blockDim.x + threadIdx.x;
  if (idx >= (long)BB * KK2 * NN) return;
  int n = (int)(idx % NN);
  long t = idx / NN;
  int k = (int)(t % KK2);
  int b = (int)(t / KK2);
  int half = k >> 8, cc = k & 255;
  const float* src = half ? xi : xr;
  float v = src[((long)b * CC + cc) * NN + n];
  X2[((long)b * NN + n) * KK2 + k] = (bf16)v;
}

// ---------------- K1: QKV GEMM  (M=N=2304, Nout=1536, K=512) ------------
__global__ __launch_bounds__(256) void qkv_gemm_kernel(const bf16* __restrict__ X2,
                                                       const bf16* __restrict__ Wcat,
                                                       const float* __restrict__ biasCat,
                                                       bf16* __restrict__ QK,
                                                       bf16* __restrict__ VT) {
  __shared__ bf16 ldsA[128 * 32];
  __shared__ bf16 ldsB[128 * 32];
  int b  = blockIdx.z;
  int tm = blockIdx.x;           // 18 tiles of 128 rows
  int tn = blockIdx.y;           // 12 tiles of 128 output cols
  int tid = threadIdx.x, lane = tid & 31, wv = tid >> 5;
  int wm = wv & 3, wn = wv >> 2;
  v8f zero = {0.f, 0.f, 0.f, 0.f, 0.f, 0.f, 0.f, 0.f};
  v8f acc[2][4];
#pragma unroll
  for (int i = 0; i < 2; ++i)
#pragma unroll
    for (int j = 0; j < 4; ++j) acc[i][j] = zero;

  for (int kk = 0; kk < KK2; kk += 32) {
    const bf16* Ag = X2   + ((long)b * NN + tm * 128) * KK2 + kk;
    const bf16* Bg = Wcat + (long)tn * 128 * KK2 + kk;
#pragma unroll
    for (int i = 0; i < 2; ++i) {
      int c = tid + i * 256;                 // 512 chunks of 8 bf16
      int row = c >> 2, co = (c & 3) * 8;
      async_copy16(&ldsA[row * 32 + co], Ag + (long)row * KK2 + co);
      async_copy16(&ldsB[row * 32 + co], Bg + (long)row * KK2 + co);
    }
    async_wait0();
    __syncthreads();
    v16bf af[2], bfm[4];
#pragma unroll
    for (int i = 0; i < 2; ++i) af[i]  = frag_a(&ldsA[(wm * 32 + i * 16) * 32], lane);
#pragma unroll
    for (int j = 0; j < 4; ++j) bfm[j] = frag_b(&ldsB[(wn * 64 + j * 16) * 32], lane);
#pragma unroll
    for (int i = 0; i < 2; ++i)
#pragma unroll
      for (int j = 0; j < 4; ++j) acc[i][j] = wmma_bf16(af[i], bfm[j], acc[i][j]);
    __syncthreads();
  }
  // epilogue: + bias, store bf16 (QK row-major; V transposed)
#pragma unroll
  for (int i = 0; i < 2; ++i)
#pragma unroll
    for (int j = 0; j < 4; ++j) {
      int row0 = tm * 128 + wm * 32 + i * 16 + (lane >> 4) * 8;
      int col  = tn * 128 + wn * 64 + j * 16 + (lane & 15);
      float bias = biasCat[col];
      v8f c = acc[i][j];
      if (col < 1024) {
#pragma unroll
        for (int r = 0; r < 8; ++r)
          QK[((long)b * NN + row0 + r) * 1024 + col] = (bf16)(c[r] + bias);
      } else {
        int vc = col - 1024;
#pragma unroll
        for (int r = 0; r < 8; ++r)
          VT[((long)b * KK2 + vc) * NN + row0 + r] = (bf16)(c[r] + bias);
      }
    }
}

// ---------------- K2: S = [qr|qi]·[kr|ki]^T / 16  (fp32 out) ------------
__global__ __launch_bounds__(256) void s_gemm_kernel(const bf16* __restrict__ QK,
                                                     float* __restrict__ S) {
  __shared__ bf16 ldsA[128 * 32];
  __shared__ bf16 ldsB[128 * 32];
  int b = blockIdx.z, tm = blockIdx.x, tn = blockIdx.y;   // 18 x 18
  int tid = threadIdx.x, lane = tid & 31, wv = tid >> 5;
  int wm = wv & 3, wn = wv >> 2;
  v8f zero = {0.f, 0.f, 0.f, 0.f, 0.f, 0.f, 0.f, 0.f};
  v8f acc[2][4];
#pragma unroll
  for (int i = 0; i < 2; ++i)
#pragma unroll
    for (int j = 0; j < 4; ++j) acc[i][j] = zero;

  for (int kk = 0; kk < KK2; kk += 32) {
    const bf16* Ag = QK + ((long)b * NN + tm * 128) * 1024 + kk;         // [qr|qi]
    const bf16* Bg = QK + ((long)b * NN + tn * 128) * 1024 + 512 + kk;   // [kr|ki]
#pragma unroll
    for (int i = 0; i < 2; ++i) {
      int c = tid + i * 256;
      int row = c >> 2, co = (c & 3) * 8;
      async_copy16(&ldsA[row * 32 + co], Ag + (long)row * 1024 + co);
      async_copy16(&ldsB[row * 32 + co], Bg + (long)row * 1024 + co);
    }
    async_wait0();
    __syncthreads();
    v16bf af[2], bfm[4];
#pragma unroll
    for (int i = 0; i < 2; ++i) af[i]  = frag_a(&ldsA[(wm * 32 + i * 16) * 32], lane);
#pragma unroll
    for (int j = 0; j < 4; ++j) bfm[j] = frag_b(&ldsB[(wn * 64 + j * 16) * 32], lane);
#pragma unroll
    for (int i = 0; i < 2; ++i)
#pragma unroll
      for (int j = 0; j < 4; ++j) acc[i][j] = wmma_bf16(af[i], bfm[j], acc[i][j]);
    __syncthreads();
  }
#pragma unroll
  for (int i = 0; i < 2; ++i)
#pragma unroll
    for (int j = 0; j < 4; ++j) {
      int row0 = tm * 128 + wm * 32 + i * 16 + (lane >> 4) * 8;
      int col  = tn * 128 + wn * 64 + j * 16 + (lane & 15);
      v8f c = acc[i][j];
#pragma unroll
      for (int r = 0; r < 8; ++r)
        S[((long)b * NN + row0 + r) * NN + col] = c[r] * 0.0625f;   // 1/sqrt(256)
    }
}

// ---------------- K3: per-row max & 1/sum(exp) --------------------------
__global__ __launch_bounds__(256) void softmax_stats_kernel(const float* __restrict__ S,
                                                            float* __restrict__ rowMax,
                                                            float* __restrict__ rowRcp) {
  long row = blockIdx.x;                 // 0 .. B*N-1
  const float* sr = S + row * NN;
  int tid = threadIdx.x;
  __shared__ float red[256];
  float mx = -1e30f;
  for (int j = tid; j < NN; j += 256) mx = fmaxf(mx, sr[j]);
  red[tid] = mx; __syncthreads();
  for (int st = 128; st > 0; st >>= 1) {
    if (tid < st) red[tid] = fmaxf(red[tid], red[tid + st]);
    __syncthreads();
  }
  mx = red[0]; __syncthreads();
  float sum = 0.f;
  for (int j = tid; j < NN; j += 256) sum += __expf(sr[j] - mx);
  red[tid] = sum; __syncthreads();
  for (int st = 128; st > 0; st >>= 1) {
    if (tid < st) red[tid] += red[tid + st];
    __syncthreads();
  }
  if (tid == 0) { rowMax[row] = mx; rowRcp[row] = 1.f / red[0]; }
}

// ---------------- K4: out = softmax(S)·V ; y_pre = x + gamma*out --------
__global__ __launch_bounds__(256) void av_gemm_kernel(const float* __restrict__ S,
                                                      const float* __restrict__ rowMax,
                                                      const float* __restrict__ rowRcp,
                                                      const bf16* __restrict__ VT,
                                                      const float* __restrict__ xr,
                                                      const float* __restrict__ xi,
                                                      const float* __restrict__ gamma,
                                                      float* __restrict__ Y) {
  __shared__ bf16 ldsA[128 * 32];
  __shared__ bf16 ldsB[128 * 32];
  int b = blockIdx.z, tm = blockIdx.x /*18*/, tn = blockIdx.y /*4*/;
  int tid = threadIdx.x, lane = tid & 31, wv = tid >> 5;
  int wm = wv & 3, wn = wv >> 2;
  float g = gamma[0];
  v8f zero = {0.f, 0.f, 0.f, 0.f, 0.f, 0.f, 0.f, 0.f};
  v8f acc[2][4];
#pragma unroll
  for (int i = 0; i < 2; ++i)
#pragma unroll
    for (int j = 0; j < 4; ++j) acc[i][j] = zero;

  for (int kk = 0; kk < NN; kk += 32) {   // 72 k-steps over keys
    // stage B first (async, runs in background of the A transform)
#pragma unroll
    for (int i = 0; i < 2; ++i) {
      int c = tid + i * 256;
      int row = c >> 2, co = (c & 3) * 8;
      const bf16* src = VT + ((long)b * KK2 + tn * 128 + row) * NN + kk + co;
      async_copy16(&ldsB[row * 32 + co], src);
    }
    // stage A: fp32 scores -> p = exp(s-m)/l -> bf16  (softmax fused here)
#pragma unroll
    for (int i = 0; i < 4; ++i) {
      int c = tid + i * 256;                 // 1024 float4 chunks
      int row = c >> 3, co = (c & 7) * 4;
      long gr = (long)b * NN + tm * 128 + row;
      float4 sv = *(const float4*)(S + gr * NN + kk + co);
      float m = rowMax[gr], rl = rowRcp[gr];
      bf16* dst = &ldsA[row * 32 + co];
      dst[0] = (bf16)(__expf(sv.x - m) * rl);
      dst[1] = (bf16)(__expf(sv.y - m) * rl);
      dst[2] = (bf16)(__expf(sv.z - m) * rl);
      dst[3] = (bf16)(__expf(sv.w - m) * rl);
    }
    async_wait0();
    __syncthreads();
    v16bf af[2], bfm[4];
#pragma unroll
    for (int i = 0; i < 2; ++i) af[i]  = frag_a(&ldsA[(wm * 32 + i * 16) * 32], lane);
#pragma unroll
    for (int j = 0; j < 4; ++j) bfm[j] = frag_b(&ldsB[(wn * 64 + j * 16) * 32], lane);
#pragma unroll
    for (int i = 0; i < 2; ++i)
#pragma unroll
      for (int j = 0; j < 4; ++j) acc[i][j] = wmma_bf16(af[i], bfm[j], acc[i][j]);
    __syncthreads();
  }
  // epilogue: y_pre = x + gamma*out  into Y[2][B][C][N]
#pragma unroll
  for (int i = 0; i < 2; ++i)
#pragma unroll
    for (int j = 0; j < 4; ++j) {
      int row0 = tm * 128 + wm * 32 + i * 16 + (lane >> 4) * 8;   // n index
      int c2   = tn * 128 + wn * 64 + j * 16 + (lane & 15);       // channel (0..511)
      int p = c2 >> 8, cch = c2 & 255;
      const float* x = p ? xi : xr;
      v8f c = acc[i][j];
#pragma unroll
      for (int r = 0; r < 8; ++r) {
        long n = row0 + r;
        float v = x[((long)b * CC + cch) * NN + n] + g * c[r];
        Y[(((long)p * BB + b) * CC + cch) * NN + n] = v;
      }
    }
}

// ---------------- K5: BN stats (mean/var -> scale/shift per channel) ----
__global__ __launch_bounds__(256) void bn_stats_kernel(const float* __restrict__ Y,
                                                       const float* __restrict__ bnwr,
                                                       const float* __restrict__ bnbr,
                                                       const float* __restrict__ bnwi,
                                                       const float* __restrict__ bnbi,
                                                       float* __restrict__ stats) {
  int ch = blockIdx.x;           // 0..511 : p = ch>>8, c = ch&255
  int p = ch >> 8, c = ch & 255;
  int tid = threadIdx.x;
  __shared__ float r1[256];
  __shared__ float r2[256];
  float s = 0.f, s2 = 0.f;
  for (int b = 0; b < BB; ++b) {
    const float* base = Y + (((long)p * BB + b) * CC + c) * NN;
    for (int n = tid; n < NN; n += 256) { float v = base[n]; s += v; s2 += v * v; }
  }
  r1[tid] = s; r2[tid] = s2; __syncthreads();
  for (int st = 128; st > 0; st >>= 1) {
    if (tid < st) { r1[tid] += r1[tid + st]; r2[tid] += r2[tid + st]; }
    __syncthreads();
  }
  if (tid == 0) {
    const float inv = 1.f / (float)(BB * NN);
    float mean = r1[0] * inv;
    float var  = r2[0] * inv - mean * mean;     // biased var
    float w  = p ? bnwi[c] : bnwr[c];
    float bb = p ? bnbi[c] : bnbr[c];
    float sc = w * rsqrtf(var + BN_EPS);
    stats[ch * 2 + 0] = sc;
    stats[ch * 2 + 1] = bb - mean * sc;
  }
}

// ---------------- K6: apply BN in place ---------------------------------
__global__ void bn_apply_kernel(float* __restrict__ Y, const float* __restrict__ stats) {
  long idx = (long)blockIdx.x * blockDim.x + threadIdx.x;
  if (idx >= 2L * BB * CC * NN) return;
  long q = idx / NN;
  int c = (int)(q % CC);
  int p = (int)(q / ((long)BB * CC));
  int key = p * CC + c;
  Y[idx] = Y[idx] * stats[key * 2] + stats[key * 2 + 1];
}

// ------------------------------------------------------------------------
extern "C" void kernel_launch(void* const* d_in, const int* in_sizes, int n_in,
                              void* d_out, int out_size, void* d_ws, size_t ws_size,
                              hipStream_t stream) {
  const float* xr   = (const float*)d_in[0];
  const float* xi   = (const float*)d_in[1];
  const float* q_wr = (const float*)d_in[2];
  const float* q_wi = (const float*)d_in[3];
  const float* q_br = (const float*)d_in[4];
  const float* q_bi = (const float*)d_in[5];
  const float* k_wr = (const float*)d_in[6];
  const float* k_wi = (const float*)d_in[7];
  const float* k_br = (const float*)d_in[8];
  const float* k_bi = (const float*)d_in[9];
  const float* v_wr = (const float*)d_in[10];
  const float* v_wi = (const float*)d_in[11];
  const float* v_br = (const float*)d_in[12];
  const float* v_bi = (const float*)d_in[13];
  const float* gamma = (const float*)d_in[14];
  const float* bn_wr = (const float*)d_in[15];
  const float* bn_br = (const float*)d_in[16];
  const float* bn_wi = (const float*)d_in[17];
  const float* bn_bi = (const float*)d_in[18];
  float* Y = (float*)d_out;

  char* ws = (char*)d_ws;
  size_t off = 0;
  auto take = [&](size_t bytes) { size_t r = off; off = (off + bytes + 255) & ~(size_t)255; return r; };
  bf16*  Wcat    = (bf16*)(ws + take((size_t)DQKV * KK2 * 2));
  float* biasCat = (float*)(ws + take((size_t)DQKV * 4));
  bf16*  X2      = (bf16*)(ws + take((size_t)BB * NN * KK2 * 2));
  bf16*  QK      = (bf16*)(ws + take((size_t)BB * NN * 1024 * 2));
  bf16*  VT      = (bf16*)(ws + take((size_t)BB * KK2 * NN * 2));
  float* S       = (float*)(ws + take((size_t)BB * NN * NN * 4));
  float* rowMax  = (float*)(ws + take((size_t)BB * NN * 4));
  float* rowRcp  = (float*)(ws + take((size_t)BB * NN * 4));
  float* stats   = (float*)(ws + take((size_t)512 * 2 * 4));

  pack_w_kernel<<<(DQKV * KK2 + 255) / 256, 256, 0, stream>>>(
      q_wr, q_wi, q_br, q_bi, k_wr, k_wi, k_br, k_bi, v_wr, v_wi, v_br, v_bi, Wcat, biasCat);

  long nx = (long)BB * KK2 * NN;
  pack_x_kernel<<<(unsigned)((nx + 255) / 256), 256, 0, stream>>>(xr, xi, X2);

  qkv_gemm_kernel<<<dim3(NN / 128, DQKV / 128, BB), 256, 0, stream>>>(X2, Wcat, biasCat, QK, VT);

  s_gemm_kernel<<<dim3(NN / 128, NN / 128, BB), 256, 0, stream>>>(QK, S);

  softmax_stats_kernel<<<BB * NN, 256, 0, stream>>>(S, rowMax, rowRcp);

  av_gemm_kernel<<<dim3(NN / 128, KK2 / 128, BB), 256, 0, stream>>>(
      S, rowMax, rowRcp, VT, xr, xi, gamma, Y);

  bn_stats_kernel<<<512, 256, 0, stream>>>(Y, bn_wr, bn_br, bn_wi, bn_bi, stats);

  long ny = 2L * BB * CC * NN;
  bn_apply_kernel<<<(unsigned)((ny + 255) / 256), 256, 0, stream>>>(Y, stats);
}
